// AttentionAugmentation2D_87522843559021
// MI455X (gfx1250) — compile-verified, matrix operand order checked by reading, not run on hardware
//
#include <hip/hip_runtime.h>

typedef __bf16 bf16;
typedef __attribute__((ext_vector_type(16))) __bf16 v16bf;
typedef __attribute__((ext_vector_type(4)))  __bf16 v4bf;
typedef __attribute__((ext_vector_type(8)))  float  v8f;
typedef __attribute__((ext_vector_type(4)))  float  f32x4;
typedef __attribute__((ext_vector_type(4)))  unsigned int u32x4;
typedef __attribute__((ext_vector_type(8)))  unsigned int u32x8;

#define NPOS       1024      // H*W = 32*32
#define DH         32        // per-head depth (dkh == dvh == 32)
#define CIN        768       // 2*DEPTH_K + DEPTH_V
#define VT_STRIDE  1032      // padded key-stride for V^T rows (16B aligned, bank-spread)
#define SPLIT      4         // workgroups per (b,h)
// LDS layout (bytes):
//   Qs  bf16[1024*32]            @      0  (65536)
//   Ks  bf16[1024*32]            @  65536  (65536)
//   Vt  bf16[32*1032]            @ 131072  (66048)   (natural key order)
//   RW  f32 [63*32]              @ 197120  ( 8064)
//   RH  f32 [63*32]              @ 205184  ( 8064)
//   Bias f32[8 waves][16*32]     @ 213248  (16384)   ([qrow][key-block])
#define SMEM_BYTES 229632

// Gather one fragment row from a row-major [rows][32] bf16 LDS tile, wave32 layout:
// lanes<16 hold K/d {0..7,16..23}, lanes>=16 hold {8..15,24..31}.  2x ds_load_b128.
__device__ __forceinline__ v16bf ld_frag(const bf16* rowbase, int hf) {
  u32x4 c0 = *(const u32x4*)(rowbase + hf * 8);
  u32x4 c1 = *(const u32x4*)(rowbase + 16 + hf * 8);
  u32x8 w = { c0.x, c0.y, c0.z, c0.w, c1.x, c1.y, c1.z, c1.w };
  return __builtin_bit_cast(v16bf, w);
}

// Exchange with the partner lane in the other 16-lane half (identity selectors).
__device__ __forceinline__ float permx16f(float x) {
  const int xi = __builtin_bit_cast(int, x);
  return __builtin_bit_cast(float, __builtin_amdgcn_permlanex16(
      xi, xi, 0x76543210, (int)0xfedcba98, false, false));
}
// max via asm: avoids the compiler's canonicalizing self-max on opaque operands.
__device__ __forceinline__ float amaxf(float a, float b) {
  asm("v_max_num_f32_e32 %0, %1, %0" : "+v"(b) : "v"(a));
  return b;
}

__global__ __launch_bounds__(256, 1)
void aa2d_flash_attn_kernel(const float* __restrict__ inp,   // [8,32,32,768]
                            const float* __restrict__ relh,  // [63,32]
                            const float* __restrict__ relw,  // [63,32]
                            float* __restrict__ out)         // [8,32,32,256]
{
  extern __shared__ char smem[];
  bf16*  Qs   = (bf16*)smem;
  bf16*  Ks   = Qs + NPOS * DH;
  bf16*  Vt   = Ks + NPOS * DH;
  float* RW   = (float*)(Vt + 32 * VT_STRIDE);
  float* RH   = RW + 63 * DH;
  float* Bias = RH + 63 * DH;

  const int blk   = blockIdx.x;
  const int pair  = blk >> 2;          // (b,h)
  const int split = blk & (SPLIT - 1); // quarter of the query tiles
  const int b     = pair >> 3;
  const int h     = pair & 7;
  const int tid   = threadIdx.x;

  // ---- stage Q (scaled, log2-domain), K, V^T into LDS as bf16 ----
  const float scale = 0.17677669529663689f * 1.4426950408889634f;  // 32^-0.5 * log2(e)
  for (int idx = tid; idx < NPOS * 8; idx += 256) {
    const int row = idx >> 3, c = idx & 7;
    const float4* base = (const float4*)(inp + ((size_t)(b * NPOS + row)) * CIN + h * DH);
    float4 q = base[c], k = base[c + 64], v = base[c + 128];
    const int d = c * 4;
    v4bf qv = { (bf16)(q.x * scale), (bf16)(q.y * scale),
                (bf16)(q.z * scale), (bf16)(q.w * scale) };
    v4bf kv = { (bf16)k.x, (bf16)k.y, (bf16)k.z, (bf16)k.w };
    *(v4bf*)(Qs + row * DH + d) = qv;   // ds_store_b64
    *(v4bf*)(Ks + row * DH + d) = kv;   // ds_store_b64
    Vt[(d + 0) * VT_STRIDE + row] = (bf16)v.x;
    Vt[(d + 1) * VT_STRIDE + row] = (bf16)v.y;
    Vt[(d + 2) * VT_STRIDE + row] = (bf16)v.z;
    Vt[(d + 3) * VT_STRIDE + row] = (bf16)v.w;
  }
  for (int i = tid; i < 63 * DH; i += 256) { RW[i] = relw[i]; RH[i] = relh[i]; }
  __syncthreads();

  // ---------------- per-wave flash attention (transposed S) ----------------
  const int wave = tid >> 5;
  const int lane = tid & 31;
  const int hf   = lane >> 4;
  const int lcol = lane & 15;        // qrow owned by this lane (both halves)
  float* biasW = Bias + wave * 512;  // [16 qrows][32 key-blocks], wave-private

  for (int tile = split * 16 + wave; tile < split * 16 + 16; tile += 8) {
    const int q0 = tile * 16;

    // Q^T B-fragment (column = query row), loaded once per tile
    v16bf Bq = ld_frag(Qs + (q0 + lcol) * DH, hf);

    // Relative-logit bias: bias[r][u] = q[n]·relw[u+c] + q[nT]·relh[u+c],
    // c = 31 - n/32, u = key-block m/32.
    const int coff = 31 - (q0 >> 5);
    for (int idx = lane; idx < 512; idx += 32) {
      const int r = idx >> 5, u = idx & 31;
      const int n  = q0 + r;
      const int nT = ((n & 31) << 5) | (n >> 5);
      const float* w  = RW + (u + coff) * DH;
      const float* hh = RH + (u + coff) * DH;
      const bf16* qn = Qs + n  * DH;
      const bf16* qt = Qs + nT * DH;
      float acc = 0.f;
#pragma unroll
      for (int d = 0; d < DH; ++d)
        acc += (float)qn[d] * w[d] + (float)qt[d] * hh[d];
      biasW[idx] = acc;
    }
    asm volatile("s_wait_dscnt 0" ::: "memory");

    float mrow = -3.0e38f, lrow = 0.f;   // per-lane scalars (row stats)
    v8f Oa = {}, Ob = {};                // O^T: d = hf*8 + r (Oa), 16 + hf*8 + r (Ob)
    const float* bptr = biasW + lcol * 32;

#pragma unroll 2
    for (int kb = 0; kb < 32; ++kb) {
      const int k0 = kb * 32;
      // K A-fragments (row = key) and V^T A-fragments (row = d, K = keys)
      v16bf Ak0 = ld_frag(Ks + (k0 + lcol) * DH, hf);
      v16bf Ak1 = ld_frag(Ks + (k0 + 16 + lcol) * DH, hf);
      v16bf Av0 = ld_frag(Vt + lcol * VT_STRIDE + k0, hf);
      v16bf Av1 = ld_frag(Vt + (16 + lcol) * VT_STRIDE + k0, hf);
      const float bv = bptr[kb];          // per-row bias, one scalar per lane

      v8f Sa = {}, Sb = {};               // S^T tiles: this lane = 16 logits of row q
      Sa = __builtin_amdgcn_wmma_f32_16x16x32_bf16(false, Ak0, false, Bq, (short)0, Sa, false, false);
      Sb = __builtin_amdgcn_wmma_f32_16x16x32_bf16(false, Ak1, false, Bq, (short)0, Sb, false, false);

      // in-lane max tree over 16 raw logits (bias folded in afterwards)
      v8f mx;
#pragma unroll
      for (int r = 0; r < 8; ++r) mx[r] = fmaxf(Sa[r], Sb[r]);
      float t = fmaxf(fmaxf(fmaxf(mx[0], mx[1]), fmaxf(mx[2], mx[3])),
                      fmaxf(fmaxf(mx[4], mx[5]), fmaxf(mx[6], mx[7])));
      t = amaxf(permx16f(t), t);          // combine with partner half
      const float nm    = amaxf(mrow, t + bv);
      const float alpha = __builtin_amdgcn_exp2f(mrow - nm);
      mrow = nm;
      const float c = nm - bv;            // exp2(s + bv - nm) == exp2(s - c)

      float p[16];
#pragma unroll
      for (int i = 0; i < 8; ++i) {
        p[i]     = __builtin_amdgcn_exp2f(Sa[i] - c);
        p[8 + i] = __builtin_amdgcn_exp2f(Sb[i] - c);
      }
      // P^T B-fragment: pure in-lane packing of S^T's C-layout (no LDS bounce)
      v16bf Pp;
#pragma unroll
      for (int i = 0; i < 16; ++i) Pp[i] = (bf16)p[i];

      float s = (((p[0] + p[1]) + (p[2] + p[3])) + ((p[4] + p[5]) + (p[6] + p[7]))) +
                (((p[8] + p[9]) + (p[10] + p[11])) + ((p[12] + p[13]) + (p[14] + p[15])));
      s += permx16f(s);
      lrow = lrow * alpha + s;
#pragma unroll
      for (int r = 0; r < 8; ++r) { Oa[r] *= alpha; Ob[r] *= alpha; }

      Oa = __builtin_amdgcn_wmma_f32_16x16x32_bf16(false, Av0, false, Pp, (short)0, Oa, false, false);
      Ob = __builtin_amdgcn_wmma_f32_16x16x32_bf16(false, Av1, false, Pp, (short)0, Ob, false, false);
    }

    // normalize and store: lane owns row q0+lcol, 8 consecutive d per fragment half
    const float inv = 1.0f / lrow;
    float* o = out + ((size_t)(b * NPOS + q0 + lcol)) * 256 + h * DH + hf * 8;
    f32x4 a0 = { Oa[0] * inv, Oa[1] * inv, Oa[2] * inv, Oa[3] * inv };
    f32x4 a1 = { Oa[4] * inv, Oa[5] * inv, Oa[6] * inv, Oa[7] * inv };
    f32x4 b0 = { Ob[0] * inv, Ob[1] * inv, Ob[2] * inv, Ob[3] * inv };
    f32x4 b1 = { Ob[4] * inv, Ob[5] * inv, Ob[6] * inv, Ob[7] * inv };
    *(f32x4*)(o)      = a0;
    *(f32x4*)(o + 4)  = a1;
    *(f32x4*)(o + 16) = b0;
    *(f32x4*)(o + 20) = b1;
  }
}

extern "C" void kernel_launch(void* const* d_in, const int* in_sizes, int n_in,
                              void* d_out, int out_size, void* d_ws, size_t ws_size,
                              hipStream_t stream) {
  (void)in_sizes; (void)n_in; (void)out_size; (void)d_ws; (void)ws_size;
  const float* inp  = (const float*)d_in[0];   // inputs
  const float* relh = (const float*)d_in[1];   // key_rel_h
  const float* relw = (const float*)d_in[2];   // key_rel_w
  float* out = (float*)d_out;

  (void)hipFuncSetAttribute((const void*)aa2d_flash_attn_kernel,
                            hipFuncAttributeMaxDynamicSharedMemorySize, SMEM_BYTES);
  aa2d_flash_attn_kernel<<<64 * SPLIT, 256, SMEM_BYTES, stream>>>(inp, relh, relw, out);
}
